// MultiHeadedTimeAttention_69595650064560
// MI455X (gfx1250) — matrix operand order repaired and checked by reading, not compiled, not run on hardware
//
#include <hip/hip_runtime.h>

// ---------------------------------------------------------------------------
// MultiHeadedTimeAttention for MI455X (gfx1250, wave32, WMMA bf16 + TDM)
// B=4, S=1024, D=768, H=12, d_k=64, n_times=2
// scores = q.(k + t_i) / sqrt(dk)  -> softmax -> @v, summed over i, then Wo.
// ---------------------------------------------------------------------------

typedef unsigned short u16;
typedef __attribute__((ext_vector_type(16))) __bf16 v16bf;
typedef __attribute__((ext_vector_type(8)))  float  v8f;
typedef __attribute__((ext_vector_type(4)))  unsigned int v4u;
typedef __attribute__((ext_vector_type(8)))  int    v8i;
typedef __attribute__((ext_vector_type(4)))  int    v4i;

#define WMMA_BF16(a, b, c) \
  __builtin_amdgcn_wmma_f32_16x16x32_bf16(false, (a), false, (b), (short)0, (c), false, false)

union BFrag   { v16bf v; u16 u[16]; uint4 q[2]; };
union BF16Bit { __bf16 h; u16 u; };

static __device__ __forceinline__ u16 f2bf(float f) { BF16Bit t; t.h = (__bf16)f; return t.u; }
static __device__ __forceinline__ float bf2f(u16 b) { BF16Bit t; t.u = b; return (float)t.h; }

// B-fragment (32x16, K x N), 16-bit: lane L -> n=L&15, hi=L>>4; element e -> K = hi*16+e.
static __device__ __forceinline__ v16bf ld_b_bf16(const u16* __restrict__ p) {
  BFrag f;
  f.q[0] = *(const uint4*)(p);
  f.q[1] = *(const uint4*)(p + 8);
  return f.v;
}

// A-fragment (16x32, M x K), 16-bit: lane L -> m=L&15, hi=L>>4;
// element e -> K = dbase + hi*16 + e.
static __device__ __forceinline__ v16bf ld_a_bf16(const u16* __restrict__ row, int dbase, int hi) {
  BFrag f;
  f.q[0] = *(const uint4*)(row + dbase + hi * 16);
  f.q[1] = *(const uint4*)(row + dbase + hi * 16 + 8);
  return f.v;
}

// A-fragment from sum of two fp32 buffers (x0 + x1 fusion for the output proj).
static __device__ __forceinline__ v16bf ld_a_f32sum(const float* __restrict__ r0,
                                                    const float* __restrict__ r1,
                                                    int k0, int hi) {
  BFrag f;
  const int o0 = k0 + hi * 16, o1 = k0 + hi * 16 + 8;
#pragma unroll
  for (int e = 0; e < 8; ++e) {
    f.u[e]     = f2bf(r0[o0 + e] + r1[o0 + e]);
    f.u[8 + e] = f2bf(r0[o1 + e] + r1[o1 + e]);
  }
  return f.v;
}

// A-fragment of probabilities from LDS score strip (already exp'ed), scaled by 1/rowsum.
static __device__ __forceinline__ v16bf ld_a_prob(const float* __restrict__ p, int hi, float s) {
  BFrag f;
  const int o0 = hi * 16, o1 = hi * 16 + 8;
#pragma unroll
  for (int e = 0; e < 8; ++e) {
    f.u[e]     = f2bf(p[o0 + e] * s);
    f.u[8 + e] = f2bf(p[o1 + e] * s);
  }
  return f.v;
}

// B-fragment of V (kp x d) from LDS stage (32 rows x 64 cols, row-major bf16).
static __device__ __forceinline__ v16bf ld_b_vlds(const u16* __restrict__ vst, int n, int hi) {
  BFrag f;
#pragma unroll
  for (int e = 0; e < 16; ++e) f.u[e] = vst[(hi * 16 + e) * 64 + n];
  return f.v;
}

// ---------------------------------------------------------------------------
// TDM: DMA a 2D tile (rows x 64 bf16, fully contiguous) from global into LDS.
// D# built per CDNA5 ISA §8.3/8.4. Wave-uniform; completion via TENSORcnt.
// ---------------------------------------------------------------------------
#if defined(__has_builtin)
#if __has_builtin(__builtin_amdgcn_tensor_load_to_lds) && \
    __has_builtin(__builtin_amdgcn_s_wait_tensorcnt)
#define HAVE_TDM 1
#endif
#endif

#ifdef HAVE_TDM
static __device__ __forceinline__ void tdm_load_rows64(const u16* gsrc, unsigned lds_off,
                                                       int rows) {
  const unsigned long long ga = (unsigned long long)(const void*)gsrc;
  // Group 0: count=1 | lds_addr | global_addr[56:0] | type=2
  v4u g0 = { 1u, lds_off, (unsigned)ga, (unsigned)((ga >> 32) & 0x01FFFFFFu) | (2u << 30) };
  // Group 1: data_size=2B; tensor 64 x rows (stride 64); tile 64 x rows.
  v8i g1 = { (int)(1u << 16),          // [17:16] data_size=1 (2 bytes)
             (int)(64u << 16),         // tensor_dim0[15:0] in [31:16]
             (int)((unsigned)rows << 16), // tensor_dim1[15:0] in [31:16]
             (int)(64u << 16),         // tile_dim0 in [31:16]
             (int)rows,                // tile_dim1 in [15:0], tile_dim2=0
             64,                       // tensor_dim0_stride[31:0]
             0,                        // stride0[47:32]=0, dim1_stride[15:0]=0
             0 };
  v4i gz = { 0, 0, 0, 0 };
#if defined(__clang_major__) && (__clang_major__ >= 23)
  v8i gz8 = { 0, 0, 0, 0, 0, 0, 0, 0 };
  __builtin_amdgcn_tensor_load_to_lds(g0, g1, gz, gz, gz8, 0);
#else
  __builtin_amdgcn_tensor_load_to_lds(g0, g1, gz, gz, 0);
#endif
}
#endif

// ---------------------------------------------------------------------------
// fp32 -> bf16 pre-conversion (weights reused 256x, activations 12x: convert
// once instead of once per consuming tile).
// ---------------------------------------------------------------------------
__global__ void cvtw_kernel(const float* __restrict__ W, u16* __restrict__ out, int n4) {
  const int i = blockIdx.x * blockDim.x + threadIdx.x;
  if (i < n4) {
    const float4 f = *(const float4*)(W + (size_t)i * 4);
    u16* o = out + (size_t)i * 4;
    o[0] = f2bf(f.x); o[1] = f2bf(f.y); o[2] = f2bf(f.z); o[3] = f2bf(f.w);
  }
}

// ---------------------------------------------------------------------------
// Projection: out[m,n] = sum_k Xb[m,k]*Wb[n,k] + bias[n]; bf16 head-split out.
// M=4096, N=768, K=768. 4 waves/WG; wave tile = 16(m) x 64(n). Xb/Wb are bf16:
// the K-loop is pure b128 loads + WMMA (no conversions).
// ---------------------------------------------------------------------------
__global__ __launch_bounds__(128) void proj_kernel(const u16* __restrict__ Xb,
                                                   const u16* __restrict__ Wb,
                                                   const float* __restrict__ bias,
                                                   u16* __restrict__ out) {
  const int L = threadIdx.x & 31, wid = threadIdx.x >> 5;
  const int n16 = L & 15, hi = L >> 4;
  const int t  = blockIdx.x * 4 + wid;   // 3072 wave tiles
  const int nt = t % 12;                 // 64-wide n tile
  const int mt = t / 12;                 // 16-wide m tile (0..255)
  const u16* Arow = Xb + (size_t)(mt * 16 + n16) * 768;
  const u16* Wrow = Wb + (size_t)(nt * 64 + n16) * 768;

  v8f c[4] = {};
  for (int kb = 0; kb < 24; ++kb) {
    const int k0 = kb * 32;
    __builtin_prefetch(Arow + k0 + 64, 0, 3);
    v16bf a = ld_a_bf16(Arow, k0, hi);
    const u16* wp = Wrow + k0 + hi * 16;
#pragma unroll
    for (int f = 0; f < 4; ++f) {
      v16bf b = ld_b_bf16(wp + (size_t)f * 16 * 768);
      c[f] = WMMA_BF16(a, b, c[f]);
    }
  }
#pragma unroll
  for (int f = 0; f < 4; ++f) {
    const int ncol = nt * 64 + f * 16 + n16;
    const float bv = bias[ncol];
    const int h = ncol >> 6, d = ncol & 63;
#pragma unroll
    for (int r = 0; r < 8; ++r) {
      const int mr = mt * 16 + r + 8 * hi;       // C layout: M = r + 8*hi
      const int b_ = mr >> 10, s = mr & 1023;
      out[(size_t)((b_ * 12 + h) * 1024 + s) * 64 + d] = f2bf(c[f][r] + bv);
    }
  }
}

// T_i += K (elementwise bf16) -> KT_i in place.
__global__ void addk_kernel(u16* __restrict__ t0, u16* __restrict__ t1,
                            const u16* __restrict__ k, int n) {
  const int i = blockIdx.x * blockDim.x + threadIdx.x;
  if (i < n) {
    const float kv = bf2f(k[i]);
    t0[i] = f2bf(bf2f(t0[i]) + kv);
    t1[i] = f2bf(bf2f(t1[i]) + kv);
  }
}

// ---------------------------------------------------------------------------
// Attention: one wave per (time, batch-head, 16-query tile). 2 waves/WG.
// Phase1: 16x1024 score strip -> LDS (stride 1028 to dodge bank conflicts).
// Phase2: row softmax (lane-pair reduce). Phase3: P@V via WMMA; V blocks are
// double-buffered TDM DMAs (issue kp+1, s_wait_tensorcnt 1, consume kp) so
// the Tensor Data Mover overlaps the matrix pipe. Buffers are wave-private,
// so the kp loop needs no workgroup barriers.
// ---------------------------------------------------------------------------
__global__ __launch_bounds__(64) void attn_kernel(const u16* __restrict__ Qh,
                                                  const u16* __restrict__ KT0,
                                                  const u16* __restrict__ KT1,
                                                  const u16* __restrict__ Vh,
                                                  float* __restrict__ x0,
                                                  float* __restrict__ x1) {
  extern __shared__ char smem[];
  const int L = threadIdx.x & 31, wid = threadIdx.x >> 5;
  const int m = L & 15, hi = L >> 4;

  float* sc   = (float*)smem + (size_t)wid * 16 * 1028;                  // 2 x 65792 B
  u16*   vst  = (u16*)(smem + 2 * 16 * 1028 * 4) + wid * 2 * 2048;       // 2 x 2 x 4096 B
  float* rinv = (float*)(smem + 2 * 16 * 1028 * 4 + 2 * 2 * 2048 * 2) + wid * 16;

  const int tid = blockIdx.x * 2 + wid;   // 6144 wave tiles
  const int qi  = tid & 63;
  const int bh  = (tid >> 6) % 48;
  const int ti  = tid / (64 * 48);
  const size_t hb = (size_t)bh * 1024 * 64;

  const u16* Qb = Qh + hb;
  const u16* KT = (ti == 0 ? KT0 : KT1) + hb;
  const u16* Vb = Vh + hb;
  float* xout   = (ti == 0 ? x0 : x1);
  const int q0  = qi * 16;

  const u16* qrow = Qb + (size_t)(q0 + m) * 64;
  const v16bf a0 = ld_a_bf16(qrow, 0, hi);    // d = 0..31
  const v16bf a1 = ld_a_bf16(qrow, 32, hi);   // d = 32..63

  // Phase 1: scores = Q @ KT^T * 1/sqrt(64)
  for (int kb = 0; kb < 64; ++kb) {
    const u16* krow = KT + (size_t)(kb * 16 + m) * 64 + hi * 16;
    v16bf b0 = ld_b_bf16(krow);
    v16bf b1 = ld_b_bf16(krow + 32);
    v8f c = {};
    c = WMMA_BF16(a0, b0, c);
    c = WMMA_BF16(a1, b1, c);
#pragma unroll
    for (int r = 0; r < 8; ++r)
      sc[(r + 8 * hi) * 1028 + kb * 16 + m] = c[r] * 0.125f;
  }
  __syncthreads();

  // Phase 2: row-wise softmax stats (2 lanes per row, wave32 shuffle combine).
  {
    const int row = L >> 1, half = L & 1;
    float* srow = sc + row * 1028;
    float mx = -3.402823466e38f;
    for (int cix = half; cix < 1024; cix += 2) mx = fmaxf(mx, srow[cix]);
    mx = fmaxf(mx, __shfl_xor(mx, 1, 32));
    float sum = 0.f;
    for (int cix = half; cix < 1024; cix += 2) {
      const float e = __expf(srow[cix] - mx);
      srow[cix] = e;
      sum += e;
    }
    sum += __shfl_xor(sum, 1, 32);
    if (half == 0) rinv[row] = 1.f / sum;
  }
  __syncthreads();

  // Phase 3: out = P @ V  (double-buffered TDM pipeline)
  const float inv = rinv[m];
  v8f o[4] = {};
#ifdef HAVE_TDM
  const unsigned vst_off = (unsigned)(unsigned long long)(const void*)vst;
  tdm_load_rows64(Vb, vst_off, 32);                       // prologue: block 0
  for (int kp = 0; kp < 32; ++kp) {
    if (kp + 1 < 32) {                                    // issue next block
      tdm_load_rows64(Vb + (size_t)(kp + 1) * 32 * 64,
                      vst_off + ((unsigned)(kp + 1) & 1u) * 4096u, 32);
      __builtin_amdgcn_s_wait_tensorcnt(1);               // block kp landed
    } else {
      __builtin_amdgcn_s_wait_tensorcnt(0);
    }
    const u16* vbuf = vst + (kp & 1) * 2048;
    v16bf pa = ld_a_prob(sc + m * 1028 + kp * 32, hi, inv);
#pragma unroll
    for (int f = 0; f < 4; ++f) {
      v16bf vb = ld_b_vlds(vbuf, f * 16 + m, hi);
      o[f] = WMMA_BF16(pa, vb, o[f]);
    }
  }
#else
  for (int kp = 0; kp < 32; ++kp) {
    const int kp0 = kp * 32;
    {  // fallback: each lane copies one 128 B row through VGPRs
      const uint4* src = (const uint4*)(Vb + (size_t)(kp0 + L) * 64);
      uint4* dst = (uint4*)(vst + L * 64);
#pragma unroll
      for (int j = 0; j < 8; ++j) dst[j] = src[j];
    }
    __syncthreads();
    v16bf pa = ld_a_prob(sc + m * 1028 + kp0, hi, inv);
#pragma unroll
    for (int f = 0; f < 4; ++f) {
      v16bf vb = ld_b_vlds(vst, f * 16 + m, hi);
      o[f] = WMMA_BF16(pa, vb, o[f]);
    }
    __syncthreads();
  }
#endif

  const int b_ = bh / 12, h = bh % 12;
#pragma unroll
  for (int f = 0; f < 4; ++f) {
    const int col = h * 64 + f * 16 + m;
#pragma unroll
    for (int r = 0; r < 8; ++r) {
      const int gm = b_ * 1024 + q0 + r + 8 * hi;
      xout[(size_t)gm * 768 + col] = o[f][r];
    }
  }
}

// ---------------------------------------------------------------------------
// Output projection: out = (x0 + x1) @ Wo^T + bo, fp32 result. Wo is bf16.
// ---------------------------------------------------------------------------
__global__ __launch_bounds__(128) void outproj_kernel(const float* __restrict__ Xa,
                                                      const float* __restrict__ Xb,
                                                      const u16* __restrict__ Wb,
                                                      const float* __restrict__ bias,
                                                      float* __restrict__ out) {
  const int L = threadIdx.x & 31, wid = threadIdx.x >> 5;
  const int n16 = L & 15, hi = L >> 4;
  const int t  = blockIdx.x * 4 + wid;
  const int nt = t % 12;
  const int mt = t / 12;
  const float* r0 = Xa + (size_t)(mt * 16 + n16) * 768;
  const float* r1 = Xb + (size_t)(mt * 16 + n16) * 768;
  const u16* Wrow = Wb + (size_t)(nt * 64 + n16) * 768;

  v8f c[4] = {};
  for (int kb = 0; kb < 24; ++kb) {
    const int k0 = kb * 32;
    v16bf a = ld_a_f32sum(r0, r1, k0, hi);
    const u16* wp = Wrow + k0 + hi * 16;
#pragma unroll
    for (int f = 0; f < 4; ++f) {
      v16bf b = ld_b_bf16(wp + (size_t)f * 16 * 768);
      c[f] = WMMA_BF16(a, b, c[f]);
    }
  }
#pragma unroll
  for (int f = 0; f < 4; ++f) {
    const int ncol = nt * 64 + f * 16 + n16;
    const float bv = bias[ncol];
#pragma unroll
    for (int r = 0; r < 8; ++r) {
      const int mr = mt * 16 + r + 8 * hi;
      out[(size_t)mr * 768 + ncol] = c[f][r] + bv;
    }
  }
}

// ---------------------------------------------------------------------------
// Host-side launch. Workspace (~69 MB):
//   Qh,Kh,Vh,T0,T1 : bf16 [B,H,S,64]  (3145728 elems each)
//   x0,x1          : f32  [B,S,768]
//   Wqb..Wob       : bf16 [768,768]   (589824 elems each)
//   Xbuf           : bf16 [4096,768]  (activation staging, reused serially)
// ---------------------------------------------------------------------------
extern "C" void kernel_launch(void* const* d_in, const int* in_sizes, int n_in,
                              void* d_out, int out_size, void* d_ws, size_t ws_size,
                              hipStream_t stream) {
  const float* query = (const float*)d_in[0];
  const float* key_  = (const float*)d_in[1];
  const float* value = (const float*)d_in[2];
  const float* times = (const float*)d_in[3];   // [2,B,S,D]
  const float* Wq = (const float*)d_in[4];
  const float* bq = (const float*)d_in[5];
  const float* Wk = (const float*)d_in[6];
  const float* bk = (const float*)d_in[7];
  const float* Wv = (const float*)d_in[8];
  const float* bv = (const float*)d_in[9];
  const float* Wt = (const float*)d_in[10];
  const float* bt = (const float*)d_in[11];
  const float* Wo = (const float*)d_in[12];
  const float* bo = (const float*)d_in[13];
  float* out = (float*)d_out;

  const size_t NE = (size_t)4 * 1024 * 768;   // 3145728 (== B*S*D == B*H*S*dk)
  const size_t NW = (size_t)768 * 768;        // 589824
  char* p = (char*)d_ws;
  u16* Qh = (u16*)p;   p += NE * 2;
  u16* Kh = (u16*)p;   p += NE * 2;
  u16* Vh = (u16*)p;   p += NE * 2;
  u16* T0 = (u16*)p;   p += NE * 2;
  u16* T1 = (u16*)p;   p += NE * 2;
  float* x0 = (float*)p; p += NE * 4;
  float* x1 = (float*)p; p += NE * 4;
  u16* Wqb = (u16*)p;  p += NW * 2;
  u16* Wkb = (u16*)p;  p += NW * 2;
  u16* Wvb = (u16*)p;  p += NW * 2;
  u16* Wtb = (u16*)p;  p += NW * 2;
  u16* Wob = (u16*)p;  p += NW * 2;
  u16* Xbuf = (u16*)p; p += NE * 2;

  // Pre-convert weights to bf16 (amortized over 256 m-tiles each).
  const unsigned cwb = (unsigned)((NW / 4 + 255) / 256);
  cvtw_kernel<<<cwb, 256, 0, stream>>>(Wq, Wqb, (int)(NW / 4));
  cvtw_kernel<<<cwb, 256, 0, stream>>>(Wk, Wkb, (int)(NW / 4));
  cvtw_kernel<<<cwb, 256, 0, stream>>>(Wv, Wvb, (int)(NW / 4));
  cvtw_kernel<<<cwb, 256, 0, stream>>>(Wt, Wtb, (int)(NW / 4));
  cvtw_kernel<<<cwb, 256, 0, stream>>>(Wo, Wob, (int)(NW / 4));

  // Each projection: convert activation to bf16 once (12x reuse), then GEMM.
  const unsigned cxb = (unsigned)((NE / 4 + 255) / 256);
  const dim3 pg(768), pb(128);
  cvtw_kernel<<<cxb, 256, 0, stream>>>(query, Xbuf, (int)(NE / 4));
  proj_kernel<<<pg, pb, 0, stream>>>(Xbuf, Wqb, bq, Qh);
  cvtw_kernel<<<cxb, 256, 0, stream>>>(key_, Xbuf, (int)(NE / 4));
  proj_kernel<<<pg, pb, 0, stream>>>(Xbuf, Wkb, bk, Kh);
  cvtw_kernel<<<cxb, 256, 0, stream>>>(value, Xbuf, (int)(NE / 4));
  proj_kernel<<<pg, pb, 0, stream>>>(Xbuf, Wvb, bv, Vh);
  cvtw_kernel<<<cxb, 256, 0, stream>>>(times, Xbuf, (int)(NE / 4));
  proj_kernel<<<pg, pb, 0, stream>>>(Xbuf, Wtb, bt, T0);
  cvtw_kernel<<<cxb, 256, 0, stream>>>(times + NE, Xbuf, (int)(NE / 4));
  proj_kernel<<<pg, pb, 0, stream>>>(Xbuf, Wtb, bt, T1);

  addk_kernel<<<(unsigned)((NE + 255) / 256), 256, 0, stream>>>(T0, T1, Kh, (int)NE);

  const size_t shbytes = 2 * 16 * 1028 * sizeof(float)   // score strips
                       + 2 * 2 * 2048 * sizeof(u16)      // double-buffered V stages
                       + 2 * 16 * sizeof(float);         // row inv-sums
  attn_kernel<<<3072, 64, shbytes, stream>>>(Qh, T0, T1, Vh, x0, x1);

  outproj_kernel<<<pg, pb, 0, stream>>>(x0, x1, Wob, bo, out);
}
